// GCN_69277822484765
// MI455X (gfx1250) — matrix operand order repaired and checked by reading, not compiled, not run on hardware
//
#include <hip/hip_runtime.h>
#include <hip/hip_bf16.h>
#include <math.h>

// Problem constants (match reference)
#define F_IN   602
#define NHID   128
#define NCLS   41
#define NCLSP  48     // N_CLS padded to 3 WMMA col-tiles
#define KS1    19     // ceil(602/32)
#define KS2    4      // 128/32

typedef __attribute__((ext_vector_type(16))) __bf16 v16bf;
typedef __attribute__((ext_vector_type(8)))  float  v8f;

__device__ __forceinline__ __bf16 bf_bits(unsigned short b) {
  __bf16 r; __builtin_memcpy(&r, &b, 2); return r;
}

// Split f32 into (bf16_hi << 16) | bf16_lo with RNE rounding; x ~= hi + lo.
__device__ __forceinline__ unsigned pack_hilo(float x) {
  union { float f; unsigned u; } c; c.f = x;
  unsigned hi = (c.u + 0x7FFFu + ((c.u >> 16) & 1u)) >> 16;
  union { unsigned u; float f; } h; h.u = hi << 16;
  union { float f; unsigned u; } l; l.f = x - h.f;
  unsigned lo = (l.u + 0x7FFFu + ((l.u >> 16) & 1u)) >> 16;
  return (hi << 16) | lo;
}

// ---------------- small elementwise kernels ----------------

__global__ void k_fill1(float* p, int n) {
  int i = blockIdx.x * blockDim.x + threadIdx.x;
  if (i < n) p[i] = 1.0f;                       // self-loop contributes 1 to in-degree
}

__global__ void k_deg(const int* __restrict__ eidx, float* deg, int E) {
  int i = blockIdx.x * blockDim.x + threadIdx.x;
  if (i < E) atomicAdd(&deg[eidx[E + i]], 1.0f);    // dst row
}

__global__ void k_rsqrt(float* p, int n) {
  int i = blockIdx.x * blockDim.x + threadIdx.x;
  if (i < n) p[i] = rsqrtf(p[i]);
}

// Pre-swizzle weights into WMMA-fragment order, split into bf16 hi/lo planes.
// Layout: [kstep][coltile][lane 0..31][j 0..15] bf16; a lane's fragment is 32
// contiguous bytes = one v16bf load in the GEMM. B 32x16 map: K = j + (lane>=16
// ? 16 : 0), N = coltile*16 + lane%16.
template<int KDIM, int KSTEPS, int NT, int NREAL>
__global__ void k_pack_wfrag(const float* __restrict__ W, int ldw,
                             __bf16* __restrict__ Wh, __bf16* __restrict__ Wl) {
  int i = blockIdx.x * blockDim.x + threadIdx.x;
  if (i >= KSTEPS * NT * 32 * 16) return;
  int j    = i & 15;
  int l    = (i >> 4) & 31;
  int rest = i >> 9;              // ks*NT + ct
  int ct   = rest % NT;
  int ks   = rest / NT;
  int k = ks * 32 + j + ((l >= 16) ? 16 : 0);
  int n = ct * 16 + (l & 15);
  float v = (k < KDIM && n < NREAL) ? W[(size_t)k * ldw + n] : 0.0f;
  unsigned p = pack_hilo(v);
  Wh[i] = bf_bits((unsigned short)(p >> 16));
  Wl[i] = bf_bits((unsigned short)(p & 0xFFFFu));
}

// out[i][f] = in[i][f]*dinv[i]^2 + bias[f]  (self-loop message + bias init)
template<int F, int FB>
__global__ void k_self_bias(const float* __restrict__ in, const float* __restrict__ dinv,
                            const float* __restrict__ bias, float* __restrict__ out, int n) {
  long long i = (long long)blockIdx.x * blockDim.x + threadIdx.x;
  if (i >= (long long)n * F) return;
  int node = (int)(i / F), f = (int)(i % F);
  float d = dinv[node];
  out[i] = in[i] * d * d + ((f < FB) ? bias[f] : 0.0f);
}

// One wave per edge: out[dst][:] += in[src][:] * dinv[src]*dinv[dst]
// float4 gathers (rows are 16B-aligned: 512B / 192B strides), scalar atomics.
template<int F>
__global__ void k_agg(const int* __restrict__ eidx, const float* __restrict__ dinv,
                      const float* __restrict__ in, float* __restrict__ out, int E) {
  int widx = (int)((blockIdx.x * (long long)blockDim.x + threadIdx.x) >> 5);
  int lane = threadIdx.x & 31;
  if (widx >= E) return;
  int s = eidx[widx], d = eidx[E + widx];
  float w = dinv[s] * dinv[d];
  const float4* ip = (const float4*)(in + (size_t)s * F);
  float* op = out + (size_t)d * F;
  #pragma unroll
  for (int q = lane; q * 4 < F; q += 32) {
    float4 v = ip[q];
    atomicAdd(&op[q * 4 + 0], v.x * w);
    atomicAdd(&op[q * 4 + 1], v.y * w);
    atomicAdd(&op[q * 4 + 2], v.z * w);
    atomicAdd(&op[q * 4 + 3], v.w * w);
  }
}

__global__ void k_relu4(float4* p, long long n4) {
  long long i = (long long)blockIdx.x * blockDim.x + threadIdx.x;
  if (i >= n4) return;
  float4 v = p[i];
  v.x = fmaxf(v.x, 0.0f); v.y = fmaxf(v.y, 0.0f);
  v.z = fmaxf(v.z, 0.0f); v.w = fmaxf(v.w, 0.0f);
  p[i] = v;
}

__global__ void k_logsoftmax(const float* __restrict__ in, float* __restrict__ out, int n) {
  int i = blockIdx.x * blockDim.x + threadIdx.x;
  if (i >= n) return;
  const float* r = in + (size_t)i * NCLSP;
  float m = -INFINITY;
  for (int c = 0; c < NCLS; ++c) m = fmaxf(m, r[c]);
  float s = 0.0f;
  for (int c = 0; c < NCLS; ++c) s += expf(r[c] - m);
  float ls = logf(s);
  float* o = out + (size_t)i * NCLS;
  for (int c = 0; c < NCLS; ++c) o[c] = r[c] - m - ls;
}

// ---------------- split-bf16 WMMA GEMM ----------------
// Out[nrows x (NW*16)] = A[nrows x KDIM] @ W[KDIM x (NW*16)]
// Block: 16 rows; wave w owns 16-col tile w. A tile staged in LDS directly in
// fragment order as separate hi/lo bf16 planes, so fragments are v16bf loads
// (2x ds_load_b128) with zero shuffle VALU. B fragments are v16bf loads from
// the pre-swizzled weight planes (L2-resident). fp32 accumulate via 3 WMMAs
// per 32-K step: Ah*Bh + Ah*Bl + Al*Bh  (~fp32 accuracy).
// Main k-loop is guard-free: OOB rows are clamped (duplicate rows only feed
// output rows we never store); the K remainder is a peeled tail step.

__device__ __forceinline__ void stage_frag_addr(int e2, int& l_row, int& ka) {
  int l = e2 >> 3;
  int j = (e2 & 7) * 2;                          // even j: ka(j+1) == ka(j)+1
  l_row = l & 15;
  ka = j + ((l >= 16) ? 16 : 0);
}

template<int KDIM, int NW>
__global__ void __launch_bounds__(NW * 32)
k_gemm_split_bf16(const float* __restrict__ A, int lda,
                  const __bf16* __restrict__ Wh, const __bf16* __restrict__ Wl,
                  float* __restrict__ Out, int nrows) {
  constexpr int KFULL = KDIM / 32;
  constexpr bool KTAIL = (KDIM % 32) != 0;
  __shared__ v16bf ash_v[32];                 // A-frag hi plane, fragment order
  __shared__ v16bf asl_v[32];                 // A-frag lo plane
  unsigned* ash32 = (unsigned*)ash_v;
  unsigned* asl32 = (unsigned*)asl_v;

  const int tid    = threadIdx.x;
  const int wave   = tid >> 5;
  const int lane   = tid & 31;
  const int laneHi = lane >> 4;
  const int l16    = lane & 15;
  const int row0   = blockIdx.x * 16;

  v8f acc = {};

  for (int ks = 0; ks < KFULL; ++ks) {
    const int k0 = ks * 32;
    for (int e2 = tid; e2 < 256; e2 += NW * 32) {
      int row, ka; stage_frag_addr(e2, row, ka);
      int gr = min(row0 + row, nrows - 1);       // clamp, no guard
      const float2 xv = *(const float2*)(A + (size_t)gr * lda + (k0 + ka));
      unsigned p0 = pack_hilo(xv.x), p1 = pack_hilo(xv.y);
      ash32[e2] = ((p1 >> 16) << 16) | (p0 >> 16);
      asl32[e2] = ((p1 & 0xFFFFu) << 16) | (p0 & 0xFFFFu);
    }
    __syncthreads();
    {
      v16bf fah = ash_v[lane];
      v16bf fal = asl_v[lane];
      const size_t wbase = ((size_t)ks * NW + wave) * 32 + lane;
      v16bf fbh = ((const v16bf*)Wh)[wbase];
      v16bf fbl = ((const v16bf*)Wl)[wbase];
      acc = __builtin_amdgcn_wmma_f32_16x16x32_bf16(false, fah, false, fbh, (short)0, acc, false, false);
      acc = __builtin_amdgcn_wmma_f32_16x16x32_bf16(false, fah, false, fbl, (short)0, acc, false, false);
      acc = __builtin_amdgcn_wmma_f32_16x16x32_bf16(false, fal, false, fbh, (short)0, acc, false, false);
    }
    __syncthreads();
  }

  if constexpr (KTAIL) {
    const int k0 = KFULL * 32;
    for (int e2 = tid; e2 < 256; e2 += NW * 32) {
      int row, ka; stage_frag_addr(e2, row, ka);
      int gr = min(row0 + row, nrows - 1);
      const float* ap = A + (size_t)gr * lda;
      int gk = k0 + ka;
      float x0 = (gk < KDIM) ? ap[gk] : 0.0f;
      float x1 = (gk + 1 < KDIM) ? ap[gk + 1] : 0.0f;
      unsigned p0 = pack_hilo(x0), p1 = pack_hilo(x1);
      ash32[e2] = ((p1 >> 16) << 16) | (p0 >> 16);
      asl32[e2] = ((p1 & 0xFFFFu) << 16) | (p0 & 0xFFFFu);
    }
    __syncthreads();
    v16bf fah = ash_v[lane];
    v16bf fal = asl_v[lane];
    const size_t wbase = ((size_t)KFULL * NW + wave) * 32 + lane;
    v16bf fbh = ((const v16bf*)Wh)[wbase];
    v16bf fbl = ((const v16bf*)Wl)[wbase];
    acc = __builtin_amdgcn_wmma_f32_16x16x32_bf16(false, fah, false, fbh, (short)0, acc, false, false);
    acc = __builtin_amdgcn_wmma_f32_16x16x32_bf16(false, fah, false, fbl, (short)0, acc, false, false);
    acc = __builtin_amdgcn_wmma_f32_16x16x32_bf16(false, fal, false, fbh, (short)0, acc, false, false);
  }

  // C/D 16x16: VGPR r -> M = r + (laneHi ? 8 : 0), N = l16
  float* orow = Out + (size_t)(row0 + laneHi * 8) * (NW * 16) + wave * 16 + l16;
  if (row0 + 16 <= nrows) {                      // uniform fast path
    #pragma unroll
    for (int r = 0; r < 8; ++r) orow[(size_t)r * (NW * 16)] = acc[r];
  } else {
    #pragma unroll
    for (int r = 0; r < 8; ++r)
      if (row0 + laneHi * 8 + r < nrows) orow[(size_t)r * (NW * 16)] = acc[r];
  }
}

// ---------------- launch ----------------

static inline size_t alignup(size_t x) { return (x + 255) & ~(size_t)255; }

extern "C" void kernel_launch(void* const* d_in, const int* in_sizes, int n_in,
                              void* d_out, int out_size, void* d_ws, size_t ws_size,
                              hipStream_t stream) {
  const float* x    = (const float*)d_in[0];
  const int*   eidx = (const int*)d_in[1];     // (2,E): [0:E)=src, [E:2E)=dst
  const float* W1   = (const float*)d_in[2];
  const float* b1   = (const float*)d_in[3];
  const float* W2   = (const float*)d_in[4];
  const float* b2   = (const float*)d_in[5];
  float* out = (float*)d_out;

  const int N = in_sizes[0] / F_IN;
  const int E = in_sizes[1] / 2;

  // Workspace layout (~103 MB; H2/H2a reuse the dead H1 region)
  const size_t w1frag = (size_t)KS1 * (NHID / 16) * 32 * 16;   // bf16 elems
  const size_t w2frag = (size_t)KS2 * (NCLSP / 16) * 32 * 16;
  char* ws = (char*)d_ws;
  size_t off = 0;
  float*  dinv = (float*)(ws + off);  off = alignup(off + (size_t)N * 4);
  __bf16* W1h  = (__bf16*)(ws + off); off = alignup(off + w1frag * 2);
  __bf16* W1l  = (__bf16*)(ws + off); off = alignup(off + w1frag * 2);
  __bf16* W2h  = (__bf16*)(ws + off); off = alignup(off + w2frag * 2);
  __bf16* W2l  = (__bf16*)(ws + off); off = alignup(off + w2frag * 2);
  size_t h1_off = off;
  float* H1  = (float*)(ws + h1_off);                                        // N x 128
  float* H1a = (float*)(ws + alignup(h1_off + (size_t)N * NHID * 4));        // N x 128
  float* H2  = H1;                                                           // N x 48 (H1 dead)
  float* H2a = (float*)(ws + alignup(h1_off + alignup((size_t)N * NCLSP * 4)));
  (void)ws_size; (void)n_in; (void)out_size;

  const int TB = 256;
  // 1) degree (init 1.0 for self-loop) -> dinv
  k_fill1<<<(N + TB - 1) / TB, TB, 0, stream>>>(dinv, N);
  k_deg<<<(E + TB - 1) / TB, TB, 0, stream>>>(eidx, dinv, E);
  k_rsqrt<<<(N + TB - 1) / TB, TB, 0, stream>>>(dinv, N);

  // 2) pre-swizzle weights into fragment-order bf16 hi/lo planes
  k_pack_wfrag<F_IN, KS1, NHID / 16, NHID>
      <<<(int)((w1frag + TB - 1) / TB), TB, 0, stream>>>(W1, NHID, W1h, W1l);
  k_pack_wfrag<NHID, KS2, NCLSP / 16, NCLS>
      <<<(int)((w2frag + TB - 1) / TB), TB, 0, stream>>>(W2, NCLS, W2h, W2l);

  const int rowBlocks = (N + 15) / 16;

  // 3) layer 1: H1 = X @ W1 (WMMA), self+bias init, edge scatter, relu
  k_gemm_split_bf16<F_IN, NHID / 16>
      <<<rowBlocks, (NHID / 16) * 32, 0, stream>>>(x, F_IN, W1h, W1l, H1, N);
  k_self_bias<NHID, NHID>
      <<<(int)(((long long)N * NHID + TB - 1) / TB), TB, 0, stream>>>(H1, dinv, b1, H1a, N);
  k_agg<NHID><<<(E + 7) / 8, TB, 0, stream>>>(eidx, dinv, H1, H1a, E);
  k_relu4<<<(int)(((long long)N * NHID / 4 + TB - 1) / TB), TB, 0, stream>>>(
      (float4*)H1a, (long long)N * NHID / 4);

  // 4) layer 2: H2 = H1a @ W2 (WMMA), self+bias init, edge scatter
  k_gemm_split_bf16<NHID, NCLSP / 16>
      <<<rowBlocks, (NCLSP / 16) * 32, 0, stream>>>(H1a, NHID, W2h, W2l, H2, N);
  k_self_bias<NCLSP, NCLS>
      <<<(int)(((long long)N * NCLSP + TB - 1) / TB), TB, 0, stream>>>(H2, dinv, b2, H2a, N);
  k_agg<NCLSP><<<(E + 7) / 8, TB, 0, stream>>>(eidx, dinv, H2, H2a, E);

  // 5) log_softmax over 41 classes -> d_out (N x 41)
  k_logsoftmax<<<(N + TB - 1) / TB, TB, 0, stream>>>(H2a, out, N);
}
